// DistributedMoE_38912403702326
// MI455X (gfx1250) — matrix operand order repaired and checked by reading, not compiled, not run on hardware
//
#include <hip/hip_runtime.h>
#include <hip/hip_bf16.h>

// ---------------------------------------------------------------------------
// MoE (top-1) with bf16x3 split-precision WMMA on gfx1250.
// Padded-permutation grouped GEMM; padding rows gather from an extra
// zero-filled row of the A matrix (single base pointer -> global_load, not flat).
// ---------------------------------------------------------------------------

typedef __attribute__((ext_vector_type(16))) __bf16 v16bf;
typedef __attribute__((ext_vector_type(8)))  __bf16 v8bf;
typedef __attribute__((ext_vector_type(8)))  float  v8f;

#define NT   16384
#define DIN  1024
#define DHID 4096
#define DOUT 1024
#define NE   8
#define NPAD (NT + NE * 128)   // 17408: worst-case 128-padded per-expert segments
#define MTILES (NPAD / 128)    // 136

union V16 { v16bf v; struct { v8bf lo; v8bf hi; } p; };

// ---------------------------------------------------------------------------
// 0) init: perm = -1 (padding sentinel), counters = 0, zero row NT of xhi/xlo
// ---------------------------------------------------------------------------
__global__ void init_kernel(int* perm, int* counts, int* cursor,
                            __bf16* xhi, __bf16* xlo) {
    int i = blockIdx.x * blockDim.x + threadIdx.x;
    if (i < NPAD) perm[i] = -1;
    if (i < NE) { counts[i] = 0; cursor[i] = 0; }
    if (i < DIN) {  // dummy gather row for padding slots
        xhi[(size_t)NT * DIN + i] = (__bf16)0.0f;
        xlo[(size_t)NT * DIN + i] = (__bf16)0.0f;
    }
}

// ---------------------------------------------------------------------------
// 1) elementwise fp32 -> (bf16 hi, bf16 lo) split (for x)
// ---------------------------------------------------------------------------
__global__ void split_kernel(const float* __restrict__ src,
                             __bf16* __restrict__ dhi, __bf16* __restrict__ dlo,
                             int n) {
    int i = blockIdx.x * blockDim.x + threadIdx.x;
    if (i < n) {
        float f = src[i];
        __bf16 h = (__bf16)f;
        dhi[i] = h;
        dlo[i] = (__bf16)(f - (float)h);
    }
}

// ---------------------------------------------------------------------------
// 2) weight transpose + split: W[e][k][n] fp32 -> Wt_{hi,lo}[e][n][k] bf16
//    (K-contiguous rows => WMMA B fragments become two b128 loads per lane)
// ---------------------------------------------------------------------------
__global__ void transpose_split_kernel(const float* __restrict__ src,
                                       __bf16* __restrict__ dhi, __bf16* __restrict__ dlo,
                                       int K, int N) {
    __shared__ float t[32][33];
    const int e  = blockIdx.z;
    const int n0 = blockIdx.x * 32;
    const int k0 = blockIdx.y * 32;
    const int tx = threadIdx.x;   // 0..31
    const int ty = threadIdx.y;   // 0..7
    const float* s = src + (size_t)e * K * N;
#pragma unroll
    for (int i = 0; i < 4; ++i) {
        int k = ty + 8 * i;
        t[k][tx] = s[(size_t)(k0 + k) * N + (n0 + tx)];
    }
    __syncthreads();
    __bf16* oh = dhi + (size_t)e * N * K;
    __bf16* ol = dlo + (size_t)e * N * K;
#pragma unroll
    for (int i = 0; i < 4; ++i) {
        int nl = ty + 8 * i;
        float f = t[tx][nl];
        __bf16 h = (__bf16)f;
        size_t o = (size_t)(n0 + nl) * K + (k0 + tx);
        oh[o] = h;
        ol[o] = (__bf16)(f - (float)h);
    }
}

// ---------------------------------------------------------------------------
// 3) router: wave-per-token, logits = x @ Wr + br, argmax (first max wins)
// ---------------------------------------------------------------------------
__global__ __launch_bounds__(256)
void router_kernel(const float* __restrict__ x, const float* __restrict__ Wr,
                   const float* __restrict__ br,
                   int* __restrict__ expert_idx, int* __restrict__ counts) {
    const int wave = threadIdx.x >> 5;
    const int lane = threadIdx.x & 31;
    const int t = blockIdx.x * 8 + wave;
    if (t >= NT) return;
    const float* xr = x + (size_t)t * DIN;
    float acc[NE];
#pragma unroll
    for (int e = 0; e < NE; ++e) acc[e] = 0.0f;
    for (int k = lane; k < DIN; k += 32) {
        float xv = xr[k];
        const float4* w = (const float4*)(Wr + (size_t)k * NE);
        float4 w0 = w[0], w1 = w[1];
        acc[0] += xv * w0.x; acc[1] += xv * w0.y;
        acc[2] += xv * w0.z; acc[3] += xv * w0.w;
        acc[4] += xv * w1.x; acc[5] += xv * w1.y;
        acc[6] += xv * w1.z; acc[7] += xv * w1.w;
    }
#pragma unroll
    for (int off = 16; off > 0; off >>= 1) {
#pragma unroll
        for (int e = 0; e < NE; ++e) acc[e] += __shfl_xor(acc[e], off, 32);
    }
    if (lane == 0) {
        int best = 0;
        float bv = acc[0] + br[0];
#pragma unroll
        for (int e = 1; e < NE; ++e) {
            float v = acc[e] + br[e];
            if (v > bv) { bv = v; best = e; }   // strict > == jnp.argmax first-max
        }
        expert_idx[t] = best;
        atomicAdd(&counts[best], 1);
    }
}

// ---------------------------------------------------------------------------
// 4) 128-aligned per-expert bases
// ---------------------------------------------------------------------------
__global__ void offsets_kernel(const int* __restrict__ counts, int* __restrict__ base) {
    if (threadIdx.x == 0 && blockIdx.x == 0) {
        int acc = 0;
        base[0] = 0;
#pragma unroll
        for (int e = 0; e < NE; ++e) {
            acc += (counts[e] + 127) & ~127;
            base[e + 1] = acc;
        }
    }
}

// ---------------------------------------------------------------------------
// 5) scatter tokens into padded permutation
// ---------------------------------------------------------------------------
__global__ void scatter_kernel(const int* __restrict__ expert_idx,
                               const int* __restrict__ base,
                               int* __restrict__ cursor, int* __restrict__ perm) {
    int t = blockIdx.x * blockDim.x + threadIdx.x;
    if (t < NT) {
        int e = expert_idx[t];
        int pos = atomicAdd(&cursor[e], 1);
        perm[base[e] + pos] = t;
    }
}

// ---------------------------------------------------------------------------
// 6) grouped GEMM, bf16x3 (hi*hi + lo*hi + hi*lo), 128x128 block tile,
//    8 waves (4 M x 2 N), each wave 2x4 WMMA tiles of 16x16, K chunked by 32.
// ---------------------------------------------------------------------------
#define SUB_M 2
#define SUB_N 4

template <bool GATHER_A, bool RELU, bool SCATTER>
__global__ __launch_bounds__(256)
void moe_gemm_kernel(const __bf16* __restrict__ Ahi, const __bf16* __restrict__ Alo,
                     const __bf16* __restrict__ Bhi, const __bf16* __restrict__ Blo,
                     const float* __restrict__ bias,
                     float* __restrict__ Cout,                 // SCATTER: [NT, N]
                     __bf16* __restrict__ Chi, __bf16* __restrict__ Clo, // !SCATTER
                     const int* __restrict__ perm, const int* __restrict__ base,
                     int K, int N) {
    __shared__ int perm_s[128];
    __shared__ int expert_s;
    const int tid = threadIdx.x;
    const int slotBase = blockIdx.y * 128;
    if (tid < 128) perm_s[tid] = perm[slotBase + tid];
    if (tid == 128) {
        int e = 0;
#pragma unroll
        for (int i = 1; i <= NE; ++i)
            if (slotBase >= base[i]) e = i;
        expert_s = (e > NE - 1) ? (NE - 1) : e;
    }
    __syncthreads();
    const int e = expert_s;

    const int wave  = tid >> 5;
    const int lane  = tid & 31;
    const int wm    = wave >> 1;      // 0..3
    const int wn    = wave & 1;       // 0..1
    const int lhalf = lane >> 4;      // 0/1
    const int l16   = lane & 15;
    const int aoff  = lhalf * 16;     // A lane K-offset within a chunk

    // A row offsets (per 16-row sub-tile; A-matrix lane M = lane&15).
    // Padding rows (perm < 0) read the zero-filled row NT: single base pointer.
    size_t arow[SUB_M];
#pragma unroll
    for (int sm = 0; sm < SUB_M; ++sm) {
        int rowL = wm * 32 + sm * 16 + l16;
        int row;
        if (GATHER_A) {
            int tok = perm_s[rowL];
            row = (tok >= 0) ? tok : NT;   // row NT is zero-filled
        } else {
            row = slotBase + rowL;
        }
        arow[sm] = (size_t)row * K;
    }

    // B column pointers (B lane: N = lane&15, K-base = 16*lhalf, contiguous)
    size_t bcol[SUB_N];
#pragma unroll
    for (int sn = 0; sn < SUB_N; ++sn) {
        int col = blockIdx.x * 128 + wn * 64 + sn * 16 + l16;
        bcol[sn] = ((size_t)e * N + col) * K + (size_t)lhalf * 16;
    }

    v8f acc[SUB_M][SUB_N];
#pragma unroll
    for (int sm = 0; sm < SUB_M; ++sm)
#pragma unroll
        for (int sn = 0; sn < SUB_N; ++sn)
            acc[sm][sn] = (v8f){0.f, 0.f, 0.f, 0.f, 0.f, 0.f, 0.f, 0.f};

    for (int kc = 0; kc < K; kc += 32) {
        V16 afh[SUB_M], afl[SUB_M];
#pragma unroll
        for (int sm = 0; sm < SUB_M; ++sm) {
            const __bf16* pah = Ahi + arow[sm] + kc + aoff;
            const __bf16* pal = Alo + arow[sm] + kc + aoff;
            afh[sm].p.lo = *(const v8bf*)(pah);
            afh[sm].p.hi = *(const v8bf*)(pah + 8);
            afl[sm].p.lo = *(const v8bf*)(pal);
            afl[sm].p.hi = *(const v8bf*)(pal + 8);
        }
        V16 bh[SUB_N], bl[SUB_N];
#pragma unroll
        for (int sn = 0; sn < SUB_N; ++sn) {
            bh[sn].v = *(const v16bf*)(Bhi + bcol[sn] + kc);
            bl[sn].v = *(const v16bf*)(Blo + bcol[sn] + kc);
        }
#pragma unroll
        for (int sm = 0; sm < SUB_M; ++sm) {
#pragma unroll
            for (int sn = 0; sn < SUB_N; ++sn) {
                acc[sm][sn] = __builtin_amdgcn_wmma_f32_16x16x32_bf16(
                    false, afh[sm].v, false, bh[sn].v, (short)0, acc[sm][sn], false, false);
                acc[sm][sn] = __builtin_amdgcn_wmma_f32_16x16x32_bf16(
                    false, afl[sm].v, false, bh[sn].v, (short)0, acc[sm][sn], false, false);
                acc[sm][sn] = __builtin_amdgcn_wmma_f32_16x16x32_bf16(
                    false, afh[sm].v, false, bl[sn].v, (short)0, acc[sm][sn], false, false);
            }
        }
    }

    // Store. C layout: VGPR r -> M = r + 8*lhalf, N = lane&15.
#pragma unroll
    for (int sn = 0; sn < SUB_N; ++sn) {
        int col = blockIdx.x * 128 + wn * 64 + sn * 16 + l16;
        float bv = bias[e * N + col];
#pragma unroll
        for (int sm = 0; sm < SUB_M; ++sm) {
#pragma unroll
            for (int r = 0; r < 8; ++r) {
                int rowL = wm * 32 + sm * 16 + lhalf * 8 + r;
                float v = acc[sm][sn][r] + bv;
                if (RELU) v = fmaxf(v, 0.0f);
                if (SCATTER) {
                    int tok = perm_s[rowL];
                    if (tok >= 0) Cout[(size_t)tok * N + col] = v;
                } else {
                    __bf16 h = (__bf16)v;
                    size_t o = (size_t)(slotBase + rowL) * N + col;
                    Chi[o] = h;
                    Clo[o] = (__bf16)(v - (float)h);
                }
            }
        }
    }
}

// ---------------------------------------------------------------------------
// launcher
// ---------------------------------------------------------------------------
extern "C" void kernel_launch(void* const* d_in, const int* in_sizes, int n_in,
                              void* d_out, int out_size, void* d_ws, size_t ws_size,
                              hipStream_t stream) {
    (void)in_sizes; (void)n_in; (void)out_size; (void)ws_size;

    const float* x  = (const float*)d_in[0];
    const float* Wr = (const float*)d_in[1];
    const float* br = (const float*)d_in[2];
    const float* W1 = (const float*)d_in[3];
    const float* b1 = (const float*)d_in[4];
    const float* W2 = (const float*)d_in[5];
    const float* b2 = (const float*)d_in[6];
    float* out = (float*)d_out;

    char* ws = (char*)d_ws;
    size_t off = 0;
    auto take = [&](size_t bytes) -> void* {
        void* p = ws + off;
        off = (off + bytes + 255) & ~(size_t)255;
        return p;
    };

    int* expert_idx = (int*)take((size_t)NT * 4);
    int* counts     = (int*)take(NE * 4);
    int* cursor     = (int*)take(NE * 4);
    int* basep      = (int*)take((NE + 1) * 4);
    int* perm       = (int*)take((size_t)NPAD * 4);

    // x split buffers have NT+1 rows: row NT is the zero dummy for padding slots
    __bf16* xhi  = (__bf16*)take((size_t)(NT + 1) * DIN * 2);
    __bf16* xlo  = (__bf16*)take((size_t)(NT + 1) * DIN * 2);
    __bf16* w1hi = (__bf16*)take((size_t)NE * DIN * DHID * 2);
    __bf16* w1lo = (__bf16*)take((size_t)NE * DIN * DHID * 2);
    __bf16* w2hi = (__bf16*)take((size_t)NE * DHID * DOUT * 2);
    __bf16* w2lo = (__bf16*)take((size_t)NE * DHID * DOUT * 2);
    __bf16* hhi  = (__bf16*)take((size_t)NPAD * DHID * 2);
    __bf16* hlo  = (__bf16*)take((size_t)NPAD * DHID * 2);

    // 0) init perm / counters / dummy zero row
    init_kernel<<<(NPAD + 255) / 256, 256, 0, stream>>>(perm, counts, cursor, xhi, xlo);

    // 1) split x
    split_kernel<<<(NT * DIN) / 256, 256, 0, stream>>>(x, xhi, xlo, NT * DIN);

    // 2) transpose + split weights: [e][k][n] -> [e][n][k]
    transpose_split_kernel<<<dim3(DHID / 32, DIN / 32, NE), dim3(32, 8), 0, stream>>>(
        W1, w1hi, w1lo, DIN, DHID);
    transpose_split_kernel<<<dim3(DOUT / 32, DHID / 32, NE), dim3(32, 8), 0, stream>>>(
        W2, w2hi, w2lo, DHID, DOUT);

    // 3) router
    router_kernel<<<NT / 8, 256, 0, stream>>>(x, Wr, br, expert_idx, counts);

    // 4) per-expert 128-aligned bases
    offsets_kernel<<<1, 32, 0, stream>>>(counts, basep);

    // 5) build padded permutation
    scatter_kernel<<<NT / 256, 256, 0, stream>>>(expert_idx, basep, cursor, perm);

    // 6) GEMM1: h = relu(gather(x) @ W1[e] + b1[e])  -> bf16 hi/lo hidden
    moe_gemm_kernel<true, true, false><<<dim3(DHID / 128, MTILES), 256, 0, stream>>>(
        xhi, xlo, w1hi, w1lo, b1, nullptr, hhi, hlo, perm, basep, DIN, DHID);

    // 7) GEMM2: out[token] = h @ W2[e] + b2[e]  (scatter by perm)
    moe_gemm_kernel<false, false, true><<<dim3(DOUT / 128, MTILES), 256, 0, stream>>>(
        hhi, hlo, w2hi, w2lo, b2, out, nullptr, nullptr, perm, basep, DHID, DOUT);
}
